// GAT_26431228740294
// MI455X (gfx1250) — matrix operand order repaired and checked by reading, not compiled, not run on hardware
//
#include <hip/hip_runtime.h>
#include <hip/hip_bf16.h>

// ---------------------------------------------------------------------------
// GAT (2-layer, PyG-style) for MI455X / gfx1250.
//  - W_dst GEMMs folded into [K,H] attention vectors (huge FLOP savings)
//  - x@W_src and h@W2_src via v_wmma_f32_16x16x32_bf16 (fragment-packed)
//  - edge softmax: 3 passes with u32-encoded float atomicMax + f32 atomicAdd
//  - scatter-aggregate: wave per edge, vectorized gather + f32 atomics (L2)
// ---------------------------------------------------------------------------

typedef __attribute__((ext_vector_type(16))) __bf16 v16bf;
typedef __attribute__((ext_vector_type(8)))  float  v8f;

static constexpr int   N_   = 50000;
static constexpr int   F_   = 128;
static constexpr int   E_   = 400000;
static constexpr int   H_   = 4;
static constexpr int   C1_  = 64;
static constexpr int   HC1_ = 256;   // H_*C1_
static constexpr int   C2_  = 64;
static constexpr int   ET_  = E_ + N_;      // edges incl. self loops
static constexpr int   MT_  = N_ / 16;      // 3125 row tiles (exact)
static constexpr float NEG_SLOPE = 0.2f;

// ---- workspace layout (bytes) ---------------------------------------------
static constexpr size_t ALN(size_t x) { return (x + 255) & ~(size_t)255; }
static constexpr size_t OFF_XPACK  = 0;                                         // bf16 N*F  (A-frag)
static constexpr size_t OFF_W1SP   = ALN(OFF_XPACK  + (size_t)N_ * F_   * 2);   // bf16 F*HC1 (B-frag)
static constexpr size_t OFF_HPACK  = ALN(OFF_W1SP   + (size_t)F_ * HC1_ * 2);   // bf16 N*HC1 (A-frag)
static constexpr size_t OFF_W2SP   = ALN(OFF_HPACK  + (size_t)N_ * HC1_ * 2);   // bf16 HC1*C2 (B-frag)
static constexpr size_t OFF_XS1    = ALN(OFF_W2SP   + (size_t)HC1_ * C2_ * 2);  // f32 N*HC1
static constexpr size_t OFF_XS2    = ALN(OFF_XS1    + (size_t)N_ * HC1_ * 4);   // f32 N*C2
static constexpr size_t OFF_HBUF   = ALN(OFF_XS2    + (size_t)N_ * C2_  * 4);   // f32 N*HC1 (layer1 agg, then relu'd h)
static constexpr size_t OFF_AS1    = ALN(OFF_HBUF   + (size_t)N_ * HC1_ * 4);   // f32 N*H
static constexpr size_t OFF_AD1    = ALN(OFF_AS1    + (size_t)N_ * H_ * 4);
static constexpr size_t OFF_M1     = ALN(OFF_AD1    + (size_t)N_ * H_ * 4);     // u32 enc max
static constexpr size_t OFF_DEN1   = ALN(OFF_M1     + (size_t)N_ * H_ * 4);
static constexpr size_t OFF_EX1    = ALN(OFF_DEN1   + (size_t)N_ * H_ * 4);     // f32 ET*H
static constexpr size_t OFF_AS2    = ALN(OFF_EX1    + (size_t)ET_ * H_ * 4);
static constexpr size_t OFF_AD2    = ALN(OFF_AS2    + (size_t)N_ * 4);
static constexpr size_t OFF_M2     = ALN(OFF_AD2    + (size_t)N_ * 4);
static constexpr size_t OFF_DEN2   = ALN(OFF_M2     + (size_t)N_ * 4);
static constexpr size_t OFF_EX2    = ALN(OFF_DEN2   + (size_t)N_ * 4);          // f32 ET
static constexpr size_t OFF_WSATT1 = ALN(OFF_EX2    + (size_t)ET_ * 4);         // f32 F*H
static constexpr size_t OFF_WDATT1 = ALN(OFF_WSATT1 + (size_t)F_ * H_ * 4);
static constexpr size_t OFF_WSATT2 = ALN(OFF_WDATT1 + (size_t)F_ * H_ * 4);     // f32 HC1
static constexpr size_t OFF_WDATT2 = ALN(OFF_WSATT2 + (size_t)HC1_ * 4);
static constexpr size_t WS_NEEDED  = ALN(OFF_WDATT2 + (size_t)HC1_ * 4);

// ---- helpers ---------------------------------------------------------------
__device__ __forceinline__ unsigned short f2bf(float f) {
  unsigned u = __float_as_uint(f);
  u += 0x7fffu + ((u >> 16) & 1u);          // round-to-nearest-even
  return (unsigned short)(u >> 16);
}
// monotone float<->u32 encoding so u32 atomicMax == float max
__device__ __forceinline__ unsigned encf(float f) {
  unsigned u = __float_as_uint(f);
  return (u & 0x80000000u) ? ~u : (u | 0x80000000u);
}
__device__ __forceinline__ float decf(unsigned e) {
  unsigned u = (e & 0x80000000u) ? (e & 0x7fffffffu) : ~e;
  return __uint_as_float(u);
}
__device__ __forceinline__ void edge_sd(const long long* __restrict__ ei, int E, int e,
                                        int& s, int& d) {
  if (e < E) { s = (int)ei[e]; d = (int)ei[(size_t)E + e]; }
  else       { s = e - E; d = s; }          // self loop
}

// ---- generic zero ----------------------------------------------------------
__global__ void k_zero(float* __restrict__ p, int n) {
  int i = blockIdx.x * blockDim.x + threadIdx.x;
  if (i < n) p[i] = 0.0f;
}

// ---- pack fp32 row-major A -> bf16 WMMA A-fragment layout ------------------
// frag half index = ((mtile*KT + kt)*32 + lane)*16 + j ; lane = 16g+row
// j<8:  K = kt*32 + 8g + j ;  j>=8:  K = kt*32 + 16 + 8g + (j-8)
__global__ void k_pack_a(const float* __restrict__ X, unsigned* __restrict__ Apack,
                         int MT, int KT, int K) {
  int id = blockIdx.x * blockDim.x + threadIdx.x;
  int total = MT * KT * 32 * 8;
  if (id >= total) return;
  int jp   = id & 7;                // pair index -> j = 2*jp (pair = consecutive K)
  int lane = (id >> 3) & 31;
  int kt   = (id >> 8) % KT;
  int mtile = id / (KT << 8);
  int g = lane >> 4, row = lane & 15;
  int j = jp * 2;
  int k = kt * 32 + (j < 8 ? 8 * g + j : 16 + 8 * g + (j - 8));
  const float* xr = X + (size_t)(mtile * 16 + row) * K + k;
  unsigned lo = f2bf(xr[0]), hi = f2bf(xr[1]);
  Apack[id] = lo | (hi << 16);
}

// ---- pack fp32 row-major W[K,Nn] -> bf16 WMMA B-fragment layout ------------
// frag half index = ((kt*NT + nt)*32 + lane)*16 + j ; col = nt*16 + lane%16 ; j<8: K = kt*32 + 8g + j ; j>=8: K = kt*32 + 16 + 8g + (j-8)
__global__ void k_pack_b(const float* __restrict__ W, unsigned short* __restrict__ Bpack,
                         int KT, int NT, int Nn) {
  int id = blockIdx.x * blockDim.x + threadIdx.x;
  int total = KT * NT * 32 * 16;
  if (id >= total) return;
  int j    = id & 15;
  int lane = (id >> 4) & 31;
  int nt   = (id >> 9) % NT;
  int kt   = id / (NT << 9);
  int g = lane >> 4;
  int n = nt * 16 + (lane & 15);
  int k = kt * 32 + (j < 8 ? 8 * g + j : 16 + 8 * g + (j - 8));
  Bpack[id] = f2bf(W[(size_t)k * Nn + n]);
}

// ---- WMMA GEMM: C[M,Nn] = Apack x Bpack, one wave per 16x16 tile -----------
__global__ void k_wmma_gemm(const unsigned short* __restrict__ Apack,
                            const unsigned short* __restrict__ Bpack,
                            float* __restrict__ C,
                            int MT, int NT, int KT, int Nn) {
  int gid  = blockIdx.x * blockDim.x + threadIdx.x;
  int wave = gid >> 5;
  int lane = threadIdx.x & 31;
  if (wave >= MT * NT) return;              // wave-uniform guard (EXEC stays all-1s)
  int mtile = wave / NT;
  int ntile = wave - mtile * NT;
  const v16bf* ap = (const v16bf*)(Apack + (size_t)mtile * KT * 32 * 16);
  const v16bf* bp = (const v16bf*)(Bpack + (size_t)ntile * 32 * 16);
  v8f acc = {};
  for (int kt = 0; kt < KT; ++kt) {
    v16bf a = ap[(size_t)kt * 32 + lane];
    v16bf b = bp[(size_t)kt * NT * 32 + lane];
    acc = __builtin_amdgcn_wmma_f32_16x16x32_bf16(false, a, false, b,
                                                  (short)0, acc, false, false);
  }
  // C/D layout: lane holds col = lane%16, rows m = 8g + r
  int g = lane >> 4, col = lane & 15;
  size_t base = (size_t)(mtile * 16 + 8 * g) * Nn + (size_t)ntile * 16 + col;
#pragma unroll
  for (int r = 0; r < 8; ++r) C[base + (size_t)r * Nn] = acc[r];
}

// ---- fold W[K,H*C] with att[H,C] -> out[K,H]  (a = X @ out) ----------------
__global__ void k_fold_att(const float* __restrict__ W, const float* __restrict__ att,
                           float* __restrict__ out, int K, int Hh, int C) {
  int id = blockIdx.x * blockDim.x + threadIdx.x;
  if (id >= K * Hh) return;
  int f = id / Hh, h = id - f * Hh;
  const float* wr = W + (size_t)f * Hh * C + (size_t)h * C;
  const float* ar = att + (size_t)h * C;
  float acc = 0.0f;
  for (int c = 0; c < C; ++c) acc = fmaf(wr[c], ar[c], acc);
  out[id] = acc;
}

// ---- GEMV: out[n,h] = X[n,:] . Wv[:,h]  ------------------------------------
__global__ void k_gemv(const float* __restrict__ X, const float* __restrict__ Wv,
                       float* __restrict__ out, int Nr, int K, int Hh) {
  int id = blockIdx.x * blockDim.x + threadIdx.x;
  if (id >= Nr * Hh) return;
  int n = id / Hh, h = id - n * Hh;
  const float* xr = X + (size_t)n * K;
  float acc = 0.0f;
  for (int f = 0; f < K; ++f) acc = fmaf(xr[f], Wv[f * Hh + h], acc);
  out[id] = acc;
}

// ---- edge pass A: segment max of leaky_relu(a_s[src]+a_d[dst]) -------------
__global__ void k_edge_max(const long long* __restrict__ ei,
                           const float* __restrict__ as, const float* __restrict__ ad,
                           unsigned* __restrict__ menc, int E, int ET, int Hh) {
  int e = blockIdx.x * blockDim.x + threadIdx.x;
  if (e >= ET) return;
  int s, d; edge_sd(ei, E, e, s, d);
  for (int h = 0; h < Hh; ++h) {
    float v = as[s * Hh + h] + ad[d * Hh + h];
    v = v > 0.0f ? v : NEG_SLOPE * v;
    atomicMax(&menc[d * Hh + h], encf(v));
  }
}

// ---- edge pass B: ex = exp(e - max[dst]); segment-sum into den -------------
__global__ void k_edge_expsum(const long long* __restrict__ ei,
                              const float* __restrict__ as, const float* __restrict__ ad,
                              const unsigned* __restrict__ menc,
                              float* __restrict__ exbuf, float* __restrict__ den,
                              int E, int ET, int Hh) {
  int e = blockIdx.x * blockDim.x + threadIdx.x;
  if (e >= ET) return;
  int s, d; edge_sd(ei, E, e, s, d);
  for (int h = 0; h < Hh; ++h) {
    float v = as[s * Hh + h] + ad[d * Hh + h];
    v = v > 0.0f ? v : NEG_SLOPE * v;
    float ex = __expf(v - decf(menc[d * Hh + h]));
    exbuf[(size_t)e * Hh + h] = ex;
    atomicAdd(&den[d * Hh + h], ex);
  }
}

// ---- edge pass C, layer1: out[dst, h*64+c] += alpha * xs[src, ...] ---------
// wave per edge; lane covers 8 contiguous feats (same head since 8 | 64)
__global__ void k_edge_scatter256(const long long* __restrict__ ei,
                                  const float* __restrict__ xs,
                                  const float* __restrict__ exbuf,
                                  const float* __restrict__ den,
                                  float* __restrict__ out, int E, int ET) {
  int gid = blockIdx.x * blockDim.x + threadIdx.x;
  int e = gid >> 5;
  if (e >= ET) return;
  int lane = threadIdx.x & 31;
  int s, d; edge_sd(ei, E, e, s, d);
  int h = lane >> 3;
  float alpha = exbuf[(size_t)e * 4 + h] / den[d * 4 + h];
  const float4* sp = (const float4*)(xs + (size_t)s * 256) + lane * 2;
  float4 v0 = sp[0], v1 = sp[1];
  float* op = out + (size_t)d * 256 + lane * 8;
  atomicAdd(op + 0, v0.x * alpha); atomicAdd(op + 1, v0.y * alpha);
  atomicAdd(op + 2, v0.z * alpha); atomicAdd(op + 3, v0.w * alpha);
  atomicAdd(op + 4, v1.x * alpha); atomicAdd(op + 5, v1.y * alpha);
  atomicAdd(op + 6, v1.z * alpha); atomicAdd(op + 7, v1.w * alpha);
}

// ---- edge pass C, layer2 (H=1, C=64): lane covers 2 feats ------------------
__global__ void k_edge_scatter64(const long long* __restrict__ ei,
                                 const float* __restrict__ xs,
                                 const float* __restrict__ exbuf,
                                 const float* __restrict__ den,
                                 float* __restrict__ out, int E, int ET) {
  int gid = blockIdx.x * blockDim.x + threadIdx.x;
  int e = gid >> 5;
  if (e >= ET) return;
  int lane = threadIdx.x & 31;
  int s, d; edge_sd(ei, E, e, s, d);
  float alpha = exbuf[e] / den[d];
  const float2* sp = (const float2*)(xs + (size_t)s * 64) + lane;
  float2 v = *sp;
  float* op = out + (size_t)d * 64 + lane * 2;
  atomicAdd(op + 0, v.x * alpha);
  atomicAdd(op + 1, v.y * alpha);
}

// ---- h = relu(agg + b1), store f32 h and bf16 A-fragment pack --------------
__global__ void k_relu_bias_pack(const float* __restrict__ agg, const float* __restrict__ bias,
                                 float* __restrict__ Hout, unsigned* __restrict__ Hpack,
                                 int MT, int KT, int K) {
  int id = blockIdx.x * blockDim.x + threadIdx.x;
  int total = MT * KT * 32 * 8;
  if (id >= total) return;
  int jp   = id & 7;
  int lane = (id >> 3) & 31;
  int kt   = (id >> 8) % KT;
  int mtile = id / (KT << 8);
  int g = lane >> 4, row = lane & 15;
  int j = jp * 2;
  int k = kt * 32 + (j < 8 ? 8 * g + j : 16 + 8 * g + (j - 8));
  size_t off = (size_t)(mtile * 16 + row) * K + k;
  float v0 = agg[off]     + bias[k];     v0 = v0 > 0.0f ? v0 : 0.0f;
  float v1 = agg[off + 1] + bias[k + 1]; v1 = v1 > 0.0f ? v1 : 0.0f;
  Hout[off]     = v0;
  Hout[off + 1] = v1;
  unsigned lo = f2bf(v0), hi = f2bf(v1);
  Hpack[id] = lo | (hi << 16);
}

// ---- out += b2 -------------------------------------------------------------
__global__ void k_add_bias(float* __restrict__ out, const float* __restrict__ b,
                           int Nr, int C) {
  int id = blockIdx.x * blockDim.x + threadIdx.x;
  if (id >= Nr * C) return;
  out[id] += b[id % C];
}

// ---------------------------------------------------------------------------
extern "C" void kernel_launch(void* const* d_in, const int* in_sizes, int n_in,
                              void* d_out, int out_size, void* d_ws, size_t ws_size,
                              hipStream_t stream) {
  if (ws_size < WS_NEEDED) return;

  const float*     x     = (const float*)d_in[0];
  const long long* ei    = (const long long*)d_in[1];
  const float*     W1s   = (const float*)d_in[2];
  const float*     W1d   = (const float*)d_in[3];
  const float*     att1s = (const float*)d_in[4];
  const float*     att1d = (const float*)d_in[5];
  const float*     b1    = (const float*)d_in[6];
  const float*     W2s   = (const float*)d_in[7];
  const float*     W2d   = (const float*)d_in[8];
  const float*     att2s = (const float*)d_in[9];
  const float*     att2d = (const float*)d_in[10];
  const float*     b2    = (const float*)d_in[11];
  float* out = (float*)d_out;
  char*  ws  = (char*)d_ws;

  unsigned*       xpack  = (unsigned*)(ws + OFF_XPACK);
  unsigned short* w1sp   = (unsigned short*)(ws + OFF_W1SP);
  unsigned*       hpack  = (unsigned*)(ws + OFF_HPACK);
  unsigned short* w2sp   = (unsigned short*)(ws + OFF_W2SP);
  float* xs1   = (float*)(ws + OFF_XS1);
  float* xs2   = (float*)(ws + OFF_XS2);
  float* hbuf  = (float*)(ws + OFF_HBUF);
  float* as1   = (float*)(ws + OFF_AS1);
  float* ad1   = (float*)(ws + OFF_AD1);
  unsigned* m1 = (unsigned*)(ws + OFF_M1);
  float* den1  = (float*)(ws + OFF_DEN1);
  float* ex1   = (float*)(ws + OFF_EX1);
  float* as2   = (float*)(ws + OFF_AS2);
  float* ad2   = (float*)(ws + OFF_AD2);
  unsigned* m2 = (unsigned*)(ws + OFF_M2);
  float* den2  = (float*)(ws + OFF_DEN2);
  float* ex2   = (float*)(ws + OFF_EX2);
  float* wsatt1 = (float*)(ws + OFF_WSATT1);
  float* wdatt1 = (float*)(ws + OFF_WDATT1);
  float* wsatt2 = (float*)(ws + OFF_WSATT2);
  float* wdatt2 = (float*)(ws + OFF_WDATT2);

  const int TB = 256;
  auto GRID = [](long long n, int tb) { return (unsigned)((n + tb - 1) / tb); };

  // --- init (encoded max = 0 == -inf; sums/agg/out = 0) ---
  k_zero<<<GRID(N_ * H_, TB), TB, 0, stream>>>((float*)m1, N_ * H_);
  k_zero<<<GRID(N_ * H_, TB), TB, 0, stream>>>(den1, N_ * H_);
  k_zero<<<GRID(N_, TB), TB, 0, stream>>>((float*)m2, N_);
  k_zero<<<GRID(N_, TB), TB, 0, stream>>>(den2, N_);
  k_zero<<<GRID((long long)N_ * HC1_, TB), TB, 0, stream>>>(hbuf, N_ * HC1_);
  k_zero<<<GRID((long long)N_ * C2_, TB), TB, 0, stream>>>(out, N_ * C2_);

  // --- folds: W_dst/W_src projected onto attention vectors ---
  k_fold_att<<<GRID(F_ * H_, TB), TB, 0, stream>>>(W1s, att1s, wsatt1, F_, H_, C1_);
  k_fold_att<<<GRID(F_ * H_, TB), TB, 0, stream>>>(W1d, att1d, wdatt1, F_, H_, C1_);
  k_fold_att<<<GRID(HC1_, TB), TB, 0, stream>>>(W2s, att2s, wsatt2, HC1_, 1, C2_);
  k_fold_att<<<GRID(HC1_, TB), TB, 0, stream>>>(W2d, att2d, wdatt2, HC1_, 1, C2_);

  // --- bf16 fragment packing ---
  k_pack_a<<<GRID((long long)MT_ * (F_ / 32) * 256, TB), TB, 0, stream>>>(
      x, xpack, MT_, F_ / 32, F_);
  k_pack_b<<<GRID((long long)(F_ / 32) * (HC1_ / 16) * 512, TB), TB, 0, stream>>>(
      W1s, w1sp, F_ / 32, HC1_ / 16, HC1_);
  k_pack_b<<<GRID((long long)(HC1_ / 32) * (C2_ / 16) * 512, TB), TB, 0, stream>>>(
      W2s, w2sp, HC1_ / 32, C2_ / 16, C2_);

  // --- layer 1: xs1 = x @ W1_src (WMMA) ; a_s1/a_d1 via folded GEMV ---
  k_wmma_gemm<<<GRID((long long)MT_ * (HC1_ / 16) * 32, TB), TB, 0, stream>>>(
      (const unsigned short*)xpack, w1sp, xs1, MT_, HC1_ / 16, F_ / 32, HC1_);
  k_gemv<<<GRID(N_ * H_, TB), TB, 0, stream>>>(x, wsatt1, as1, N_, F_, H_);
  k_gemv<<<GRID(N_ * H_, TB), TB, 0, stream>>>(x, wdatt1, ad1, N_, F_, H_);

  // --- layer 1 edge softmax + aggregate ---
  k_edge_max<<<GRID(ET_, TB), TB, 0, stream>>>(ei, as1, ad1, m1, E_, ET_, H_);
  k_edge_expsum<<<GRID(ET_, TB), TB, 0, stream>>>(ei, as1, ad1, m1, ex1, den1, E_, ET_, H_);
  k_edge_scatter256<<<GRID((long long)ET_ * 32, TB), TB, 0, stream>>>(
      ei, xs1, ex1, den1, hbuf, E_, ET_);

  // --- h = relu(agg + b1); pack bf16 fragments (in place over hbuf f32) ---
  k_relu_bias_pack<<<GRID((long long)MT_ * (HC1_ / 32) * 256, TB), TB, 0, stream>>>(
      hbuf, b1, hbuf, hpack, MT_, HC1_ / 32, HC1_);

  // --- layer 2 ---
  k_wmma_gemm<<<GRID(((long long)MT_ * (C2_ / 16) * 32 + TB - 1), TB), TB, 0, stream>>>(
      (const unsigned short*)hpack, w2sp, xs2, MT_, C2_ / 16, HC1_ / 32, C2_);
  k_gemv<<<GRID(N_, TB), TB, 0, stream>>>(hbuf, wsatt2, as2, N_, HC1_, 1);
  k_gemv<<<GRID(N_, TB), TB, 0, stream>>>(hbuf, wdatt2, ad2, N_, HC1_, 1);

  k_edge_max<<<GRID(ET_, TB), TB, 0, stream>>>(ei, as2, ad2, m2, E_, ET_, 1);
  k_edge_expsum<<<GRID(ET_, TB), TB, 0, stream>>>(ei, as2, ad2, m2, ex2, den2, E_, ET_, 1);
  k_edge_scatter64<<<GRID((long long)ET_ * 32, TB), TB, 0, stream>>>(
      ei, xs2, ex2, den2, out, E_, ET_);

  k_add_bias<<<GRID((long long)N_ * C2_, TB), TB, 0, stream>>>(out, b2, N_, C2_);
}
